// SelfAttentionLayer_32710470926953
// MI455X (gfx1250) — matrix operand order repaired and checked by reading, not compiled, not run on hardware
//
#include <hip/hip_runtime.h>

// Flash attention (causal), S=8192, D=128, fp32 in/out, bf16 WMMA compute.
// gfx1250 / CDNA5: wave32, v_wmma_f32_16x16x32_bf16, async global->LDS pipeline,
// DPP16 row reductions, 64-key chunks, per-chunk specialized straight-line paths
// (hot path: no masks, no inner branches), Q fragments parked in LDS.

#define SEQ 8192
#define HD  128
#define BQ  64                 // query rows per block
#define BK  64                 // key rows per chunk (2x WMMA K)
#define NWAVE (BQ / 16)        // 4 waves
#define NTHREADS (NWAVE * 32)  // 128 threads
#define KPAD (HD + 16)         // sK row stride (elems): 288B -> 32B-aligned v16bf loads
#define VPAD (BK + 16)         // sVt row stride (elems): 160B -> 32B-aligned v16bf loads
#define CHUNK_FLOATS (BK * HD)                         // 8192 floats = 32KB per tensor
#define VEC4_PER_THREAD (CHUNK_FLOATS / 4 / NTHREADS)  // 16

// Dynamic LDS partition (bytes)
#define OFF_STAGEK 0
#define OFF_STAGEV (OFF_STAGEK + CHUNK_FLOATS * 4)           // 32768
#define OFF_SK     (OFF_STAGEV + CHUNK_FLOATS * 4)           // 65536
#define OFF_SVT    (OFF_SK + BK * KPAD * 2)                  // + 18432
#define OFF_SP     (OFF_SVT + HD * VPAD * 2)                 // + 20480
#define OFF_QF     (OFF_SP + NWAVE * 16 * BK * 2)            // + 8192
#define SMEM_BYTES (OFF_QF + NWAVE * 4 * 32 * 32)            // + 16384 = 129024

typedef __attribute__((ext_vector_type(16))) __bf16 v16bf;
typedef __attribute__((ext_vector_type(8)))  float  v8f;

// A-fragment (16-bit, 16x32) element->K mapping (contiguous, matches B loads):
// lanes 0-15: K={0..15}; lanes 16-31: K={16..31}.
__device__ __forceinline__ int a_frag_k(int e, int g) {
    return e + g * 16;
}

// Async DMA of one 16B packet: global -> LDS, tracked by ASYNCcnt.
__device__ __forceinline__ void async_copy_b128(const void* gsrc, void* ldst) {
    asm volatile("global_load_async_to_lds_b128 %0, %1, off"
                 :: "v"((unsigned)(uintptr_t)ldst), "v"(gsrc)
                 : "memory");
}

__device__ __forceinline__ void wait_async0() {
#if __has_builtin(__builtin_amdgcn_s_wait_asynccnt)
    __builtin_amdgcn_s_wait_asynccnt(0);
#else
    asm volatile("s_wait_asynccnt 0x0" ::: "memory");
#endif
}

__device__ __forceinline__ float fast_exp2(float x) {
#if __has_builtin(__builtin_amdgcn_exp2f)
    return __builtin_amdgcn_exp2f(x);
#else
    return exp2f(x);
#endif
}

// DPP16 lane-xor within each 16-lane row (EXEC all-ones in compute region).
template <int CTRL>
__device__ __forceinline__ float dpp_xor(float x) {
    int v = __builtin_amdgcn_update_dpp(0, __builtin_bit_cast(int, x),
                                        CTRL, 0xF, 0xF, true);
    return __builtin_bit_cast(float, v);
}
// xor1 = quad_perm(1,0,3,2)=0xB1; xor2 = quad_perm(2,3,0,1)=0x4E;
// xor4 = row_half_mirror=0x141;   xor8 = row_mirror=0x140.
__device__ __forceinline__ float rowred_max(float x) {
    x = fmaxf(x, dpp_xor<0xB1>(x));
    x = fmaxf(x, dpp_xor<0x4E>(x));
    x = fmaxf(x, dpp_xor<0x141>(x));
    x = fmaxf(x, dpp_xor<0x140>(x));
    return x;
}
__device__ __forceinline__ float rowred_sum(float x) {
    x += dpp_xor<0xB1>(x);
    x += dpp_xor<0x4E>(x);
    x += dpp_xor<0x141>(x);
    x += dpp_xor<0x140>(x);
    return x;
}

// One 16-query x 64-key flash-attention step, specialized on the number of
// live 16-key tiles (NT) and whether causal masking is needed (DIAG).
// Straight-line: the scheduler can batch ds_loads deep ahead of the WMMAs.
template <int NT, bool DIAG>
__device__ __forceinline__ void process_chunk(
    int k0, int qRow0, int laneRow, int g, int waveU,
    const __bf16 (*sK)[KPAD], const __bf16 (*sVt)[VPAD],
    __bf16 (*sP)[16][BK], const __bf16* qBase,
    v8f* acc, float* m_run, float* l_run)
{
    v8f zero;
    #pragma unroll
    for (int r = 0; r < 8; ++r) zero[r] = 0.0f;   // folds to inline C=0

    // ---- S = (Q*scale).K^T : f outer so only one Q fragment lives ----
    v8f s[NT];
    #pragma unroll
    for (int f = 0; f < 4; ++f) {
        v16bf qa = *(const v16bf*)(qBase + f * 512);
        #pragma unroll
        for (int t = 0; t < NT; ++t) {
            v16bf b = *(const v16bf*)(&sK[t * 16 + laneRow][f * 32 + g * 16]);
            s[t] = __builtin_amdgcn_wmma_f32_16x16x32_bf16(
                       false, qa, false, b, (short)0,
                       (f == 0 ? zero : s[t]), false, false);
        }
    }

    if (DIAG) {  // elementwise causal mask (diagonal chunk only)
        #pragma unroll
        for (int t = 0; t < NT; ++t) {
            #pragma unroll
            for (int r = 0; r < 8; ++r) {
                int qAbs = qRow0 + r + 8 * g;
                if (k0 + t * 16 + laneRow > qAbs) s[t][r] = -3.0e38f;
            }
        }
    }

    // ---- row max over live tiles + DPP16 xor-reduce ----
    v8f mx = s[0];
    #pragma unroll
    for (int t = 1; t < NT; ++t)
        #pragma unroll
        for (int r = 0; r < 8; ++r) mx[r] = fmaxf(mx[r], s[t][r]);
    #pragma unroll
    for (int r = 0; r < 8; ++r) mx[r] = rowred_max(mx[r]);

    // ---- online softmax update (base-2, native v_exp_f32) ----
    float alpha[8];
    #pragma unroll
    for (int r = 0; r < 8; ++r) {
        float mnew = fmaxf(m_run[r], mx[r]);
        alpha[r]  = fast_exp2(m_run[r] - mnew);
        m_run[r]  = mnew;
    }
    v8f psum;
    #pragma unroll
    for (int r = 0; r < 8; ++r) psum[r] = 0.0f;
    #pragma unroll
    for (int t = 0; t < NT; ++t)
        #pragma unroll
        for (int r = 0; r < 8; ++r) {
            s[t][r] = fast_exp2(s[t][r] - m_run[r]);   // P in place
            psum[r] += s[t][r];
        }
    #pragma unroll
    for (int r = 0; r < 8; ++r)
        l_run[r] = l_run[r] * alpha[r] + rowred_sum(psum[r]);
    #pragma unroll
    for (int dt = 0; dt < 8; ++dt)
        #pragma unroll
        for (int r = 0; r < 8; ++r) acc[dt][r] *= alpha[r];

    // ---- P: C/D layout -> A layout via per-wave LDS staging ----
    #pragma unroll
    for (int t = 0; t < NT; ++t)
        #pragma unroll
        for (int r = 0; r < 8; ++r)
            sP[waveU][r + 8 * g][t * 16 + laneRow] = (__bf16)s[t][r];
    if (NT & 1) {  // dead odd tile inside a live half: its P == 0
        #pragma unroll
        for (int r = 0; r < 8; ++r)
            sP[waveU][r + 8 * g][NT * 16 + laneRow] = (__bf16)0.0f;
    }
    asm volatile("" ::: "memory");   // same-wave LDS ops are in-order

    // ---- O += P.V : 8 WMMAs per live 32-key half ----
    constexpr int NH = (NT + 1) / 2;
    #pragma unroll
    for (int h = 0; h < NH; ++h) {
        v16bf pa;
        #pragma unroll
        for (int e = 0; e < 16; ++e)
            pa[e] = sP[waveU][laneRow][a_frag_k(e, g) + h * 32];
        #pragma unroll
        for (int dt = 0; dt < 8; ++dt) {
            v16bf bv = *(const v16bf*)(&sVt[dt * 16 + laneRow][h * 32 + g * 16]);
            acc[dt] = __builtin_amdgcn_wmma_f32_16x16x32_bf16(
                          false, pa, false, bv, (short)0, acc[dt], false, false);
        }
    }
}

__launch_bounds__(NTHREADS, 1)
__global__ void fa_causal_kernel(const float* __restrict__ q,
                                 const float* __restrict__ k,
                                 const float* __restrict__ v,
                                 float* __restrict__ out) {
    extern __shared__ __align__(32) char smem[];
    float* sStageK = (float*)(smem + OFF_STAGEK);             // raw fp32 K (async dst)
    float* sStageV = (float*)(smem + OFF_STAGEV);             // raw fp32 V (async dst)
    __bf16 (*sK)[KPAD]   = (__bf16(*)[KPAD])(smem + OFF_SK);  // K bf16 row-major
    __bf16 (*sVt)[VPAD]  = (__bf16(*)[VPAD])(smem + OFF_SVT); // V bf16 transposed [d][key]
    __bf16 (*sP)[16][BK] = (__bf16(*)[16][BK])(smem + OFF_SP);// per-wave P staging
    __bf16* qLds         = (__bf16*)(smem + OFF_QF);          // Q A-fragments, per lane

    const int tid     = threadIdx.x;
    const int lane    = tid & 31;
    const int laneRow = lane & 15;
    const int g       = lane >> 4;             // lane group (wave32 halves)
    const int qBlock  = blockIdx.x * BQ;
    // Scalar wave id: all causal guards become pure SALU compares + branches.
    const int waveU   = __builtin_amdgcn_readfirstlane(tid) >> 5;
    const int qRow0   = qBlock + waveU * 16;   // scalar
    // scale = 1/sqrt(128) with log2(e) folded in: softmax runs on native exp2.
    const float qscale = 0.08838834764831845f * 1.4426950408889634f;

    // ---- Q tile -> bf16 A-fragments, parked in LDS (frees 32 VGPRs) ----
    __bf16* qBase = qLds + (size_t)(waveU * 4 * 32 + lane) * 16;  // frag f at +f*512
    {
        const float* qrow = q + (size_t)(qRow0 + laneRow) * HD;
        #pragma unroll
        for (int f = 0; f < 4; ++f) {
            v16bf qa;
            #pragma unroll
            for (int e = 0; e < 16; ++e)
                qa[e] = (__bf16)(qrow[a_frag_k(e, g) + f * 32] * qscale);
            *(v16bf*)(qBase + f * 512) = qa;
        }
    }

    // ---- Online-softmax state (C/D layout: elem r <-> row r + 8*g) ----
    v8f acc[8];
    float m_run[8], l_run[8];
    #pragma unroll
    for (int dt = 0; dt < 8; ++dt)
        #pragma unroll
        for (int r = 0; r < 8; ++r) acc[dt][r] = 0.0f;
    #pragma unroll
    for (int r = 0; r < 8; ++r) { m_run[r] = -3.0e38f; l_run[r] = 0.0f; }

    // Issue async DMA for one 64x128 K/V chunk into the fp32 staging buffers.
    auto issue_chunk = [&](int k0) {
        const float* kg = k + (size_t)k0 * HD;
        const float* vg = v + (size_t)k0 * HD;
        #pragma unroll
        for (int i = 0; i < VEC4_PER_THREAD; ++i) {
            int idx = tid + i * NTHREADS;        // 0..2047, coalesced
            async_copy_b128(kg + idx * 4, sStageK + idx * 4);
            async_copy_b128(vg + idx * 4, sStageV + idx * 4);
        }
    };

    const int numChunks = (qBlock + BQ) / BK;    // blockIdx.x + 1
    issue_chunk(0);                              // prime the pipeline

    for (int c = 0; c < numChunks; ++c) {
        const int k0 = c * BK;

        // Wait for our async writes; barrier makes all waves' writes visible and
        // proves the previous chunk's compute is done (bf16 tiles reusable).
        wait_async0();
        __syncthreads();

        // ---- LDS-local convert: fp32 stage -> bf16 tiles (K row-major, V transposed)
        #pragma unroll
        for (int i = 0; i < VEC4_PER_THREAD; ++i) {
            int idx = tid + i * NTHREADS;
            int row = idx >> 5;                  // 32 float4 per 128-wide row
            int c4  = (idx & 31) * 4;
            const float4 kv = *(const float4*)(sStageK + idx * 4);
            union { __bf16 h[4]; uint2 u; } pk;
            pk.h[0] = (__bf16)kv.x; pk.h[1] = (__bf16)kv.y;
            pk.h[2] = (__bf16)kv.z; pk.h[3] = (__bf16)kv.w;
            *(uint2*)(&sK[row][c4]) = pk.u;
            const float4 vv = *(const float4*)(sStageV + idx * 4);
            sVt[c4 + 0][row] = (__bf16)vv.x;
            sVt[c4 + 1][row] = (__bf16)vv.y;
            sVt[c4 + 2][row] = (__bf16)vv.z;
            sVt[c4 + 3][row] = (__bf16)vv.w;
        }
        __syncthreads();   // bf16 tiles ready; staging buffers free again

        // Kick off DMA for the NEXT chunk; it overlaps with this chunk's WMMAs.
        if (c + 1 < numChunks) issue_chunk(k0 + BK);

        // Per-chunk dispatch (all-scalar conditions -> s_cmp/s_cbranch):
        // non-diagonal chunks take the branch-free, mask-free hot path.
        if (k0 <= qRow0 + 15) {
            if (k0 + BK - 1 <= qRow0) {
                process_chunk<4, false>(k0, qRow0, laneRow, g, waveU,
                                        sK, sVt, sP, qBase, acc, m_run, l_run);
            } else if (k0 + 48 <= qRow0 + 15) {
                process_chunk<4, true>(k0, qRow0, laneRow, g, waveU,
                                       sK, sVt, sP, qBase, acc, m_run, l_run);
            } else if (k0 + 32 <= qRow0 + 15) {
                process_chunk<3, true>(k0, qRow0, laneRow, g, waveU,
                                       sK, sVt, sP, qBase, acc, m_run, l_run);
            } else if (k0 + 16 <= qRow0 + 15) {
                process_chunk<2, true>(k0, qRow0, laneRow, g, waveU,
                                       sK, sVt, sP, qBase, acc, m_run, l_run);
            } else {
                process_chunk<1, true>(k0, qRow0, laneRow, g, waveU,
                                       sK, sVt, sP, qBase, acc, m_run, l_run);
            }
        }
    }

    // ---- normalize and write fp32 output ----
    #pragma unroll
    for (int r = 0; r < 8; ++r) {
        float inv = 1.0f / l_run[r];
        size_t rowOff = (size_t)(qRow0 + r + 8 * g) * HD;
        #pragma unroll
        for (int dt = 0; dt < 8; ++dt)
            out[rowOff + dt * 16 + laneRow] = acc[dt][r] * inv;
    }
}

extern "C" void kernel_launch(void* const* d_in, const int* in_sizes, int n_in,
                              void* d_out, int out_size, void* d_ws, size_t ws_size,
                              hipStream_t stream) {
    const float* q = (const float*)d_in[0];
    const float* k = (const float*)d_in[1];
    const float* v = (const float*)d_in[2];
    float* out = (float*)d_out;
    (void)in_sizes; (void)n_in; (void)out_size; (void)d_ws; (void)ws_size;

    // Allow >64KB dynamic LDS (WGP has 320KB). Host-side, graph-capture safe.
    (void)hipFuncSetAttribute(reinterpret_cast<const void*>(fa_causal_kernel),
                              hipFuncAttributeMaxDynamicSharedMemorySize,
                              SMEM_BYTES);

    dim3 grid(SEQ / BQ);        // 128 blocks
    dim3 block(NTHREADS);       // 4 waves (wave32)
    hipLaunchKernelGGL(fa_causal_kernel, grid, block, SMEM_BYTES, stream,
                       q, k, v, out);
}
